// GraphModel_4561255269155
// MI455X (gfx1250) — matrix operand order repaired and checked by reading, not compiled, not run on hardware
//
#include <hip/hip_runtime.h>
#include <math.h>

// ---------------------------------------------------------------------------
// Sizes from the reference
// ---------------------------------------------------------------------------
#define Bn     16
#define Mn     256
#define Dn     64
#define Hn     8
#define STEPSn 20
#define Tn     4
#define Nn     (Bn * Mn)          // 4096 nodes
#define ROWSn  (Nn * Tn)          // 16384 decoder rows
#define LN_EPS 1e-5f

typedef _Float16 f16;
typedef __attribute__((ext_vector_type(8)))  _Float16 v8h;
typedef __attribute__((ext_vector_type(16))) _Float16 v16h;
typedef __attribute__((ext_vector_type(8)))  float    v8f;
typedef __attribute__((ext_vector_type(4)))  int      v4i;

#define DEV static __device__ __forceinline__

DEV v16h cat8(v8h lo, v8h hi) {
  return __builtin_shufflevector(lo, hi, 0,1,2,3,4,5,6,7,8,9,10,11,12,13,14,15);
}

// A fragment (16x32 f16, row-major source, 'stride' halves per row).
// lane<16: row=lane,    K in {kb..kb+7, kb+16..kb+23}
// lane>=16: row=lane-16, K in {kb+8..kb+15, kb+24..kb+31}
DEV v16h loadAfrag(const f16* base, int stride, int kb, int lane) {
  const f16* p = base + (size_t)(lane & 15) * stride + (lane >> 4) * 8 + kb;
  return cat8(*(const v8h*)p, *(const v8h*)(p + 16));
}

// B fragment (32x16 f16) from B^T stored row-major (row = output column).
// lane<16: col=lane, K in {kb..kb+7, kb+16..kb+23} ; lane>=16: col=lane-16, K in {kb+8..kb+15, kb+24..kb+31}
DEV v16h loadBfrag(const f16* baseT, int stride, int kb, int lane) {
  const f16* p = baseT + (size_t)(lane & 15) * stride + (lane >> 4) * 8 + kb;
  return cat8(*(const v8h*)p, *(const v8h*)(p + 16));
}

DEV v8f wmma16(v16h a, v16h b, v8f c) {
  return __builtin_amdgcn_wmma_f32_16x16x32_f16(false, a, false, b, (short)0, c,
                                                false, false);
}

DEV v8f splat8(float v) {
  v8f r;
  #pragma unroll
  for (int i = 0; i < 8; ++i) r[i] = v;
  return r;
}

// 16-byte global -> LDS async copy (ASYNCcnt path); plain-copy fallback.
// Builtin takes generic v4i* pointers: (dst_lds, src_global, imm off, imm cpol)
DEV void asyncCopy16(const f16* g, f16* l) {
#if __has_builtin(__builtin_amdgcn_global_load_async_to_lds_b128)
  __builtin_amdgcn_global_load_async_to_lds_b128((v4i*)l, (v4i*)g, 0, 0);
#else
  *(v8h*)l = *(const v8h*)g;
#endif
}

DEV void waitAsync() {
#if __has_builtin(__builtin_amdgcn_s_wait_asynccnt)
  __builtin_amdgcn_s_wait_asynccnt(0);
#else
  asm volatile("s_wait_asynccnt 0x0" ::: "memory");
#endif
}

// ---------------------------------------------------------------------------
// Prep: transpose weights to f16 B^T layout, fold edge-MLP to 4 scalars
// ---------------------------------------------------------------------------
__global__ void prep_kernel(const float* asw, const float* asb,
                            const float* adw, const float* adb,
                            const float* aew, const float* aeb,
                            const float* dw1, const float* dw2,
                            f16* W16, float* biasAll, f16* dW1T, f16* dW2T,
                            float* consts) {
  int tid = threadIdx.x;
  for (int i = tid; i < 16 * Dn * Dn; i += 256) {   // W16[mat][fo][k]
    int mat = i >> 12, fo = (i >> 6) & 63, k = i & 63;
    float v = (mat < 8) ? asw[(size_t)mat * 4096 + k * 64 + fo]
                        : adw[(size_t)(mat - 8) * 4096 + k * 64 + fo];
    W16[i] = (f16)v;
  }
  for (int i = tid; i < 16 * Dn; i += 256) {
    int mat = i >> 6, j = i & 63;
    biasAll[i] = (mat < 8) ? asb[mat * 64 + j] : adb[(mat - 8) * 64 + j];
  }
  for (int i = tid; i < Dn * Dn; i += 256) {        // decoder W^T
    int fo = i >> 6, k = i & 63;
    dW1T[i] = (f16)dw1[k * 64 + fo];
    dW2T[i] = (f16)dw2[k * 64 + fo];
  }
  if (tid < 3) { float s = 0.f; for (int j = 0; j < 64; ++j) s += aew[tid*64+j]; consts[tid] = s; }
  if (tid == 3) { float s = 0.f; for (int j = 0; j < 64; ++j) s += aeb[j]; consts[3] = s; }
}

// ---------------------------------------------------------------------------
// Encoder: 3x (Linear -> LN [-> ReLU]) ; 4 nodes per 256-thread block
// ---------------------------------------------------------------------------
__global__ void encoder_kernel(const float* x, const float* w1, const float* b1,
                               const float* w2, const float* b2,
                               const float* w3, const float* b3,
                               const float* g1, const float* be1,
                               const float* g2, const float* be2,
                               const float* g3, const float* be3,
                               float* nf32, f16* nf16) {
  __shared__ float sh[4][Dn];
  int tid = threadIdx.x, g = tid >> 6, f = tid & 63;
  int n = blockIdx.x * 4 + g;

  // layer 1: [20]->[64]
  float acc = b1[f];
  #pragma unroll
  for (int k = 0; k < STEPSn; ++k) acc += x[(size_t)n * STEPSn + k] * w1[k * 64 + f];
  sh[g][f] = acc; __syncthreads();
  float s1 = 0.f, s2 = 0.f;
  for (int j = 0; j < 64; ++j) { float v = sh[g][j]; s1 += v; s2 += v * v; }
  float mu = s1 * (1.f/64.f), var = s2 * (1.f/64.f) - mu * mu;
  float h = fmaxf((acc - mu) * rsqrtf(var + LN_EPS) * g1[f] + be1[f], 0.f);
  __syncthreads(); sh[g][f] = h; __syncthreads();

  // layer 2
  acc = b2[f];
  for (int k = 0; k < 64; ++k) acc += sh[g][k] * w2[k * 64 + f];
  __syncthreads(); sh[g][f] = acc; __syncthreads();
  s1 = 0.f; s2 = 0.f;
  for (int j = 0; j < 64; ++j) { float v = sh[g][j]; s1 += v; s2 += v * v; }
  mu = s1 * (1.f/64.f); var = s2 * (1.f/64.f) - mu * mu;
  h = fmaxf((acc - mu) * rsqrtf(var + LN_EPS) * g2[f] + be2[f], 0.f);
  __syncthreads(); sh[g][f] = h; __syncthreads();

  // layer 3 (LN, no relu)
  acc = b3[f];
  for (int k = 0; k < 64; ++k) acc += sh[g][k] * w3[k * 64 + f];
  __syncthreads(); sh[g][f] = acc; __syncthreads();
  s1 = 0.f; s2 = 0.f;
  for (int j = 0; j < 64; ++j) { float v = sh[g][j]; s1 += v; s2 += v * v; }
  mu = s1 * (1.f/64.f); var = s2 * (1.f/64.f) - mu * mu;
  float o = (acc - mu) * rsqrtf(var + LN_EPS) * g3[f] + be3[f];
  nf32[(size_t)n * 64 + f] = o;
  nf16[(size_t)n * 64 + f] = (f16)o;
}

// ---------------------------------------------------------------------------
// Generic WMMA GEMM: OUT[r,0:64] = A16[r,:64] @ W + bias  (W passed as W^T)
// one wave -> one 16-row tile, all four 16-col tiles (A fragments reused)
// Templated on output type: no branches in the store path.
// ---------------------------------------------------------------------------
template <bool F16OUT>
__global__ void __launch_bounds__(256)
gemm_rows64(const f16* __restrict__ A, const f16* __restrict__ WT,
            const float* __restrict__ bias, void* __restrict__ outp, int R,
            size_t wtStride, size_t biasStride, size_t outStride) {
  int wid = threadIdx.x >> 5, lane = threadIdx.x & 31;
  int gw = blockIdx.x * 8 + wid;
  if (gw * 16 >= R) return;
  int mat = blockIdx.y;
  const f16* Abase = A + (size_t)gw * 16 * 64;
  v16h a0 = loadAfrag(Abase, 64, 0, lane);
  v16h a1 = loadAfrag(Abase, 64, 32, lane);
  const f16* WTm = WT + (size_t)mat * wtStride;
  const float* bm = bias + (size_t)mat * biasStride;
  #pragma unroll
  for (int ct = 0; ct < 4; ++ct) {
    int col = ct * 16 + (lane & 15);
    const f16* Bb = WTm + (size_t)ct * 16 * 64;
    // two independent accumulator chains, combined at the end
    v8f c0 = wmma16(a0, loadBfrag(Bb, 64, 0, lane), splat8(bm[col]));
    v8f c1 = wmma16(a1, loadBfrag(Bb, 64, 32, lane), splat8(0.f));
    v8f acc = c0 + c1;
    #pragma unroll
    for (int r = 0; r < 8; ++r) {
      size_t row = (size_t)gw * 16 + r + (lane >> 4) * 8;
      size_t o = (size_t)mat * outStride + row * 64 + col;
      if (F16OUT) ((f16*)outp)[o] = (f16)acc[r];
      else        ((float*)outp)[o] = acc[r];
    }
  }
}

// ---------------------------------------------------------------------------
// Scores + softmax + head-mean -> K = A - I   (one block per (b, 16-row tile))
// S[s,d] = (Hi[s]·Hj[d] + edge_score(s,d)) / 8 ; diagonal of A_raw stays 0
// ---------------------------------------------------------------------------
__global__ void __launch_bounds__(256)
scores_kernel(const f16* __restrict__ HiHj, const float* __restrict__ coord,
              const float* __restrict__ ew, const float* __restrict__ consts,
              f16* __restrict__ Kmat) {
  __shared__ float S[16 * 256];
  __shared__ float Asum[16 * 256];
  __shared__ float Es[16 * 256];
  __shared__ float red[16 * 16];
  __shared__ float rowv[16];

  int tid = threadIdx.x;
  int b = blockIdx.x >> 4, stile = blockIdx.x & 15;
  int wid = tid >> 5, lane = tid & 31;
  int sl = tid >> 4, lid = tid & 15;
  int sg = stile * 16 + sl, gs = b * Mn + sg;

  float cx = consts[0], cy = consts[1], cw = consts[2], cb = consts[3];
  #pragma unroll
  for (int j = 0; j < 16; ++j) {
    int d = j * 16 + lid;
    float es = 0.f;
    if (d != sg) {
      int gd = b * Mn + d;
      int e = b * (Mn * (Mn - 1)) + sg * (Mn - 1) + d - (d > sg ? 1 : 0);
      es = cx * (coord[2*gd] - coord[2*gs]) + cy * (coord[2*gd+1] - coord[2*gs+1])
         + cw * ew[e] + cb;
    }
    Es[sl * 256 + d] = es;
    Asum[sl * 256 + d] = 0.f;
  }

  for (int h = 0; h < Hn; ++h) {
    // --- 16x256 raw dot tile via WMMA: A = Hi rows, B = Hj rows (col-major) ---
    const f16* Hi = HiHj + ((size_t)h * Nn + b * Mn + stile * 16) * 64;
    v16h a0 = loadAfrag(Hi, 64, 0, lane);
    v16h a1 = loadAfrag(Hi, 64, 32, lane);
    v8f acc[2];
    int dts[2] = {wid * 2, wid * 2 + 1};
    #pragma unroll
    for (int j = 0; j < 2; ++j) {
      const f16* Hj = HiHj + ((size_t)(8 + h) * Nn + b * Mn + dts[j] * 16) * 64;
      v8f c0 = wmma16(a0, loadBfrag(Hj, 64, 0, lane), splat8(0.f));
      v8f c1 = wmma16(a1, loadBfrag(Hj, 64, 32, lane), splat8(0.f));
      acc[j] = c0 + c1;
    }
    __syncthreads();           // previous head's element phase fully done
    #pragma unroll
    for (int j = 0; j < 2; ++j) {
      #pragma unroll
      for (int r = 0; r < 8; ++r)
        S[(r + (lane >> 4) * 8) * 256 + dts[j] * 16 + (lane & 15)] = acc[j][r];
    }
    __syncthreads();

    // --- scale + edge score + zero diagonal, then row softmax over 256 ---
    float rm = -1e30f;
    #pragma unroll
    for (int j = 0; j < 16; ++j) {
      int d = j * 16 + lid;
      float v = S[sl * 256 + d];
      v = (d == sg) ? 0.f : (v + Es[sl * 256 + d]) * 0.125f;
      S[sl * 256 + d] = v;
      rm = fmaxf(rm, v);
    }
    red[sl * 16 + lid] = rm; __syncthreads();
    if (lid == 0) { float m = red[sl*16]; for (int t = 1; t < 16; ++t) m = fmaxf(m, red[sl*16+t]); rowv[sl] = m; }
    __syncthreads();
    float m = rowv[sl], ss = 0.f;
    #pragma unroll
    for (int j = 0; j < 16; ++j) {
      int d = j * 16 + lid;
      float e = __expf(S[sl * 256 + d] - m);
      S[sl * 256 + d] = e; ss += e;
    }
    red[sl * 16 + lid] = ss; __syncthreads();
    if (lid == 0) { float s = 0.f; for (int t = 0; t < 16; ++t) s += red[sl*16+t]; rowv[sl] = s; }
    __syncthreads();
    float inv = 1.0f / rowv[sl];
    #pragma unroll
    for (int j = 0; j < 16; ++j) {
      int d = j * 16 + lid;
      Asum[sl * 256 + d] += S[sl * 256 + d] * inv;
    }
  }
  __syncthreads();
  #pragma unroll
  for (int j = 0; j < 16; ++j) {
    int d = j * 16 + lid;
    float kv = Asum[sl * 256 + d] * 0.125f - (d == sg ? 1.f : 0.f);
    Kmat[((size_t)(b * Mn + sg)) * 256 + d] = (f16)kv;
  }
}

// ---------------------------------------------------------------------------
// ODE: kout[b] = K[b] (256x256) @ yin[b] (256x64), yin given transposed f16.
// All 8 waves of a block share batch b: stage the whole 64x256 B operand in
// LDS via async global->LDS copies, then feed WMMA B fragments from LDS.
// ---------------------------------------------------------------------------
__global__ void __launch_bounds__(256)
ode_gemm(const f16* __restrict__ Kmat, const f16* __restrict__ yinT,
         float* __restrict__ kout) {
  __shared__ f16 Bs[Dn * Mn];                   // 32 KB
  int tid = threadIdx.x, wid = tid >> 5, lane = tid & 31;
  int gw = blockIdx.x * 8 + wid;                // 1024 tiles total
  int b = blockIdx.x >> 3;                      // 8 waves per block share b
  int rem = gw & 63, stile = rem >> 2, ft = rem & 3;

  // async-stage yinT[b] (64 rows x 256) into LDS, 16B per thread per iter
  const f16* Bg = yinT + (size_t)b * Dn * Mn;
  #pragma unroll
  for (int it = 0; it < 8; ++it) {
    int chunk = it * 256 + tid;                 // 8-half chunks
    asyncCopy16(Bg + chunk * 8, &Bs[chunk * 8]);
  }
  const f16* Ab = Kmat + ((size_t)b * Mn + stile * 16) * 256;
  __builtin_prefetch(Ab, 0, 3);                 // global_prefetch_b8
  waitAsync();
  __syncthreads();

  const f16* Bb = Bs + (size_t)ft * 16 * 256;
  v8f c0 = splat8(0.f), c1 = splat8(0.f);
  #pragma unroll
  for (int kk = 0; kk < 8; kk += 2) {
    c0 = wmma16(loadAfrag(Ab, 256, kk * 32, lane),
                loadBfrag(Bb, 256, kk * 32, lane), c0);
    c1 = wmma16(loadAfrag(Ab, 256, (kk + 1) * 32, lane),
                loadBfrag(Bb, 256, (kk + 1) * 32, lane), c1);
  }
  v8f acc = c0 + c1;
  int col = ft * 16 + (lane & 15);
  #pragma unroll
  for (int r = 0; r < 8; ++r) {
    size_t row = (size_t)b * Mn + stile * 16 + r + (lane >> 4) * 8;
    kout[row * 64 + col] = acc[r];
  }
}

__global__ void init_y(const float* nf32, float* y, f16* z16) {
  int idx = blockIdx.x * 256 + threadIdx.x;     // N*D = 262144
  int n = idx >> 6, f = idx & 63;
  float v = nf32[idx];
  y[idx] = v;
  z16[((size_t)n * Tn + 0) * 64 + f] = (f16)v;  // traj t=0
}

__global__ void combine_yin(const float* y, const float* k1, const float* k2,
                            const float* k3, float a0, float a1, float a2,
                            float a3, f16* yinT) {
  int idx = blockIdx.x * 256 + threadIdx.x;
  int n = idx >> 6, f = idx & 63, b = n >> 8, m = n & 255;
  float v = a0 * y[idx] + a1 * k1[idx] + a2 * k2[idx] + a3 * k3[idx];
  yinT[((size_t)b * Dn + f) * 256 + m] = (f16)v;
}

__global__ void update_y(float* y, const float* k1, const float* k2,
                         const float* k3, const float* k4, float dt, int step,
                         f16* z16) {
  int idx = blockIdx.x * 256 + threadIdx.x;
  int n = idx >> 6, f = idx & 63;
  float v = y[idx] + (k1[idx] + 3.f * (k2[idx] + k3[idx]) + k4[idx]) * dt * 0.125f;
  y[idx] = v;
  z16[((size_t)n * Tn + step + 1) * 64 + f] = (f16)v;
}

// ---------------------------------------------------------------------------
// Decoder LN (+ReLU): one wave per 64-wide row, wave32 shuffle reduction
// ---------------------------------------------------------------------------
__global__ void ln_relu_kernel(const float* in, const float* g, const float* be,
                               f16* out, int R) {
  int wid = threadIdx.x >> 5, lane = threadIdx.x & 31;
  int row = blockIdx.x * 8 + wid;
  if (row >= R) return;
  float v0 = in[(size_t)row * 64 + lane];
  float v1 = in[(size_t)row * 64 + lane + 32];
  float s1 = v0 + v1, s2 = v0 * v0 + v1 * v1;
  #pragma unroll
  for (int off = 16; off > 0; off >>= 1) {
    s1 += __shfl_xor(s1, off, 32);
    s2 += __shfl_xor(s2, off, 32);
  }
  float mu = s1 * (1.f/64.f), var = s2 * (1.f/64.f) - mu * mu;
  float inv = rsqrtf(var + LN_EPS);
  float o0 = fmaxf((v0 - mu) * inv * g[lane] + be[lane], 0.f);
  float o1 = fmaxf((v1 - mu) * inv * g[lane + 32] + be[lane + 32], 0.f);
  out[(size_t)row * 64 + lane]      = (f16)o0;
  out[(size_t)row * 64 + lane + 32] = (f16)o1;
}

__global__ void final_dot(const f16* __restrict__ z, const float* __restrict__ w3,
                          const float* __restrict__ b3, float* __restrict__ out) {
  int r = blockIdx.x * 256 + threadIdx.x;       // 16384 outputs
  float acc = b3[0];
  const v8h* zp = (const v8h*)(z + (size_t)r * 64);
  #pragma unroll
  for (int c = 0; c < 8; ++c) {
    v8h v = zp[c];
    #pragma unroll
    for (int i = 0; i < 8; ++i) acc += (float)v[i] * w3[c * 8 + i];
  }
  out[r] = acc;
}

// ---------------------------------------------------------------------------
// Host orchestration
// ---------------------------------------------------------------------------
extern "C" void kernel_launch(void* const* d_in, const int* in_sizes, int n_in,
                              void* d_out, int out_size, void* d_ws, size_t ws_size,
                              hipStream_t stream) {
  const float* x      = (const float*)d_in[0];
  const float* coord  = (const float*)d_in[1];
  const float* ew     = (const float*)d_in[3];
  const float* enc_w1 = (const float*)d_in[5],  *enc_b1 = (const float*)d_in[6];
  const float* enc_w2 = (const float*)d_in[7],  *enc_b2 = (const float*)d_in[8];
  const float* enc_w3 = (const float*)d_in[9],  *enc_b3 = (const float*)d_in[10];
  const float* enc_g1 = (const float*)d_in[11], *enc_be1 = (const float*)d_in[12];
  const float* enc_g2 = (const float*)d_in[13], *enc_be2 = (const float*)d_in[14];
  const float* enc_g3 = (const float*)d_in[15], *enc_be3 = (const float*)d_in[16];
  const float* asw = (const float*)d_in[17], *asb = (const float*)d_in[18];
  const float* adw = (const float*)d_in[19], *adb = (const float*)d_in[20];
  const float* aew = (const float*)d_in[21], *aeb = (const float*)d_in[22];
  const float* dw1 = (const float*)d_in[23], *db1 = (const float*)d_in[24];
  const float* dw2 = (const float*)d_in[25], *db2 = (const float*)d_in[26];
  const float* dw3 = (const float*)d_in[27], *db3 = (const float*)d_in[28];
  const float* dg1 = (const float*)d_in[29], *dbe1 = (const float*)d_in[30];
  const float* dg2 = (const float*)d_in[31], *dbe2 = (const float*)d_in[32];
  float* out = (float*)d_out;

  // workspace bump allocator (256B aligned)
  char* ws = (char*)d_ws;
  size_t off = 0;
  auto alloc = [&](size_t bytes) { char* p = ws + off; off = (off + bytes + 255) & ~(size_t)255; return p; };
  float* consts  = (float*)alloc(256);
  float* nf32    = (float*)alloc((size_t)Nn * Dn * 4);
  f16*   nf16    = (f16*)  alloc((size_t)Nn * Dn * 2);
  f16*   W16     = (f16*)  alloc((size_t)16 * Dn * Dn * 2);
  float* biasAll = (float*)alloc((size_t)16 * Dn * 4);
  f16*   dW1T    = (f16*)  alloc((size_t)Dn * Dn * 2);
  f16*   dW2T    = (f16*)  alloc((size_t)Dn * Dn * 2);
  f16*   HiHj    = (f16*)  alloc((size_t)16 * Nn * Dn * 2);      // 8 MB
  f16*   Kmat    = (f16*)  alloc((size_t)Bn * Mn * Mn * 2);      // 2 MB
  float* y       = (float*)alloc((size_t)Nn * Dn * 4);
  float* k1      = (float*)alloc((size_t)Nn * Dn * 4);
  float* k2      = (float*)alloc((size_t)Nn * Dn * 4);
  float* k3      = (float*)alloc((size_t)Nn * Dn * 4);
  float* k4      = (float*)alloc((size_t)Nn * Dn * 4);
  f16*   yinT    = (f16*)  alloc((size_t)Nn * Dn * 2);
  f16*   z16     = (f16*)  alloc((size_t)ROWSn * Dn * 2);        // 2 MB
  float* gout    = (float*)alloc((size_t)ROWSn * Dn * 4);        // 4 MB
  f16*   lf16    = (f16*)  alloc((size_t)ROWSn * Dn * 2);        // 2 MB
  (void)ws_size; (void)in_sizes; (void)n_in; (void)out_size;

  prep_kernel<<<1, 256, 0, stream>>>(asw, asb, adw, adb, aew, aeb, dw1, dw2,
                                     W16, biasAll, dW1T, dW2T, consts);
  encoder_kernel<<<Nn / 4, 256, 0, stream>>>(x, enc_w1, enc_b1, enc_w2, enc_b2,
                                             enc_w3, enc_b3, enc_g1, enc_be1,
                                             enc_g2, enc_be2, enc_g3, enc_be3,
                                             nf32, nf16);
  // 16 per-head projections: HiHj[mat] = nf @ W[mat] + b[mat]   (f16 out)
  gemm_rows64<true><<<dim3(Nn / 16 / 8, 16), 256, 0, stream>>>(
      nf16, W16, biasAll, HiHj, Nn,
      (size_t)Dn * Dn, (size_t)Dn, (size_t)Nn * Dn);
  scores_kernel<<<Bn * 16, 256, 0, stream>>>(HiHj, coord, ew, consts, Kmat);

  const int ND_BLKS = (Nn * Dn) / 256;          // 1024
  init_y<<<ND_BLKS, 256, 0, stream>>>(nf32, y, z16);
  const float dt = 1.0f / 3.0f;
  for (int step = 0; step < Tn - 1; ++step) {
    combine_yin<<<ND_BLKS, 256, 0, stream>>>(y, k1, k2, k3, 1.f, 0.f, 0.f, 0.f, yinT);
    ode_gemm<<<128, 256, 0, stream>>>(Kmat, yinT, k1);
    combine_yin<<<ND_BLKS, 256, 0, stream>>>(y, k1, k2, k3, 1.f, dt / 3.f, 0.f, 0.f, yinT);
    ode_gemm<<<128, 256, 0, stream>>>(Kmat, yinT, k2);
    combine_yin<<<ND_BLKS, 256, 0, stream>>>(y, k1, k2, k3, 1.f, -dt / 3.f, dt, 0.f, yinT);
    ode_gemm<<<128, 256, 0, stream>>>(Kmat, yinT, k3);
    combine_yin<<<ND_BLKS, 256, 0, stream>>>(y, k1, k2, k3, 1.f, dt, -dt, dt, yinT);
    ode_gemm<<<128, 256, 0, stream>>>(Kmat, yinT, k4);
    update_y<<<ND_BLKS, 256, 0, stream>>>(y, k1, k2, k3, k4, dt, step, z16);
  }

  // decoder (f32 GEMM outputs feed the LN kernels)
  gemm_rows64<false><<<dim3(ROWSn / 16 / 8, 1), 256, 0, stream>>>(
      z16, dW1T, db1, gout, ROWSn, 0, 0, 0);
  ln_relu_kernel<<<ROWSn / 8, 256, 0, stream>>>(gout, dg1, dbe1, lf16, ROWSn);
  gemm_rows64<false><<<dim3(ROWSn / 16 / 8, 1), 256, 0, stream>>>(
      lf16, dW2T, db2, gout, ROWSn, 0, 0, 0);
  ln_relu_kernel<<<ROWSn / 8, 256, 0, stream>>>(gout, dg2, dbe2, lf16, ROWSn);
  final_dot<<<ROWSn / 256, 256, 0, stream>>>(lf16, dw3, db3, out);
}
